// CausalSelfAttention_28028956573842
// MI455X (gfx1250) — matrix operand order repaired and checked by reading, not compiled, not run on hardware
//
#include <hip/hip_runtime.h>
#include <math.h>

// ---------------------------------------------------------------------------
// MI455X / gfx1250 implementation of the chunked state+SWA attention layer.
// Matrix math uses v_wmma_f32_16x16x32_bf16 (wave32 WMMA); GEMM tile staging
// uses the Tensor Data Mover (tensor_load_to_lds + s_wait_tensorcnt).
// ---------------------------------------------------------------------------

typedef __bf16 bf16;
typedef __attribute__((ext_vector_type(16))) __bf16 v16bf;
typedef __attribute__((ext_vector_type(8)))  float  v8f;
typedef unsigned int u32;
typedef u32 __attribute__((ext_vector_type(4))) uint32x4;
typedef int __attribute__((ext_vector_type(4))) int32x4;
typedef int __attribute__((ext_vector_type(8))) int32x8;

#define Bc      4
#define Tc      2048
#define Cc      1024
#define Hc      16
#define Nc      64
#define CHUNKc  256
#define SSTATEc 256
#define SINKc   1
#define PARTIALc 32
#define NEGc    (-1e30f)
#define Mrows   (Bc*Tc)

#if defined(__has_builtin)
#  if __has_builtin(__builtin_amdgcn_tensor_load_to_lds) && __has_builtin(__builtin_amdgcn_s_wait_tensorcnt)
#    define HAVE_TDM 1
#  else
#    define HAVE_TDM 0
#  endif
#else
#  define HAVE_TDM 0
#endif

#if HAVE_TDM
#  warning "CDNA5 path probe: TDM enabled (tensor_load_to_lds + s_wait_tensorcnt)"
#else
#  warning "CDNA5 path probe: TDM builtin unavailable, cooperative-load fallback"
#endif

__device__ __forceinline__ float waveSum(float v){
  for (int o = 16; o > 0; o >>= 1) v += __shfl_xor(v, o, 32);
  return v;
}
__device__ __forceinline__ float waveMax(float v){
  for (int o = 16; o > 0; o >>= 1) v = fmaxf(v, __shfl_xor(v, o, 32));
  return v;
}
__device__ __forceinline__ bf16 tobf(float f){ return (bf16)f; }

#if HAVE_TDM
// Issue a 2-D TDM load: tile_d1 rows x tile_d0 elements of 4-byte data,
// row stride stride0 (elements), from global gaddr into LDS byte offset lds_off.
// Tensor dims == tile dims so every element is in-bounds.
__device__ __forceinline__ void tdm_load_2d_f32(u32 lds_off, const void* gaddr,
                                                u32 stride0, u32 tile_d0, u32 tile_d1)
{
  unsigned long long ga = (unsigned long long)gaddr;
  uint32x4 g0;
  g0[0] = 1u;                                   // count=1 (valid), user mode
  g0[1] = lds_off;                              // lds_addr (bytes)
  g0[2] = (u32)(ga & 0xffffffffu);              // global_addr[31:0]
  g0[3] = (u32)((ga >> 32) & 0x01ffffffu)       // global_addr[56:32]
          | 0x80000000u;                        // type=2 ("image")
  int32x8 g1;
  g1[0] = 0x00020000;                           // data_size=2 (4 bytes)
  g1[1] = (int)((tile_d0 & 0xffffu) << 16);     // tensor_dim0[15:0] (== tile_d0)
  g1[2] = (int)(((tile_d0 >> 16) & 0xffffu) | ((tile_d1 & 0xffffu) << 16)); // td0 hi | td1 lo
  g1[3] = (int)(((tile_d1 >> 16) & 0xffffu) | ((tile_d0 & 0xffffu) << 16)); // td1 hi | tile_dim0
  g1[4] = (int)(tile_d1 & 0xffffu);             // tile_dim1 | tile_dim2=0
  g1[5] = (int)stride0;                         // tensor_dim0_stride[31:0]
  g1[6] = 0;                                    // stride0[47:32] | stride1[15:0]
  g1[7] = 0;
  int32x4 z4 = {0, 0, 0, 0};
#if __clang_major__ >= 23
  int32x8 z8 = {0, 0, 0, 0, 0, 0, 0, 0};
  __builtin_amdgcn_tensor_load_to_lds(g0, g1, z4, z4, z8, 0);
#else
  __builtin_amdgcn_tensor_load_to_lds(g0, g1, z4, z4, 0);
#endif
}
#endif

// ---------------------------------------------------------------------------
// Generic WMMA GEMM:  D[m,n] = sum_k A[m,k] * W[n,k]  (+ add[m,n] if add!=0)
// Block: 128 threads (4 waves). Block tile 64x64, wave tile 16x64.
// Tiles staged in LDS as f32 via TDM, converted to bf16 at fragment build.
// ---------------------------------------------------------------------------
__global__ void gemm_wmma(const float* __restrict__ A, const float* __restrict__ W,
                          const float* __restrict__ add, float* __restrict__ D,
                          int M, int Nn, int K)
{
  __shared__ float fA[64][32];
  __shared__ float fB[64][32];
  const int m0 = blockIdx.x * 64;
  const int n0 = blockIdx.y * 64;
  const int tid  = threadIdx.x;
  const int wid  = tid >> 5;
  const int lane = tid & 31;

  v8f acc[4] = {};

#if HAVE_TDM
  // LDS aperture: low 32 bits of a flat shared pointer are the LDS byte offset.
  const u32 offA = (u32)(size_t)&fA[0][0];
  const u32 offB = (u32)(size_t)&fB[0][0];
#endif

  for (int kb = 0; kb < K; kb += 32) {
#if HAVE_TDM
    if (wid == 0) {
      tdm_load_2d_f32(offA, &A[(size_t)m0 * K + kb], (u32)K, 32u, 64u);
      tdm_load_2d_f32(offB, &W[(size_t)n0 * K + kb], (u32)K, 32u, 64u);
    }
    __builtin_amdgcn_s_wait_tensorcnt(0);
#else
    for (int i = tid; i < 64*32; i += 128) {
      int r = i >> 5, c = i & 31;
      fA[r][c] = A[(size_t)(m0 + r)*K + kb + c];
      fB[r][c] = W[(size_t)(n0 + r)*K + kb + c];
    }
#endif
    if (kb + 32 < K) {
      __builtin_prefetch(&A[(size_t)(m0 + (tid >> 1))*K + kb + 32], 0, 1);
      __builtin_prefetch(&W[(size_t)(n0 + (tid >> 1))*K + kb + 32], 0, 1);
    }
    __syncthreads();

    // A fragment (16x32 bf16)
    v16bf af;
    {
      int ar  = wid*16 + (lane & 15);
      int kb0 = (lane < 16) ? 0 : 16;
      for (int j = 0; j < 8; ++j)
        for (int p = 0; p < 2; ++p) {
          int kk = kb0 + 2*j + p;
          af[2*j + p] = tobf(fA[ar][kk]);
        }
    }
    int kbb = (lane < 16) ? 0 : 16;
    for (int nt = 0; nt < 4; ++nt) {
      v16bf bfr;
      int bc = nt*16 + (lane & 15);
      for (int j = 0; j < 8; ++j)
        for (int p = 0; p < 2; ++p)
          bfr[2*j + p] = tobf(fB[bc][kbb + 2*j + p]);
      acc[nt] = __builtin_amdgcn_wmma_f32_16x16x32_bf16(false, af, false, bfr,
                                                        (short)0, acc[nt], false, false);
    }
    __syncthreads();
  }

  const int col = lane & 15;
  const int rb  = (lane < 16) ? 0 : 8;
  for (int nt = 0; nt < 4; ++nt)
    for (int r = 0; r < 8; ++r) {
      size_t m = (size_t)m0 + wid*16 + rb + r;
      size_t n = (size_t)n0 + nt*16 + col;
      float v = acc[nt][r];
      if (add) v += add[m*Nn + n];
      D[m*Nn + n] = v;
    }
}

// ---------------------------------------------------------------------------
// Per-(b,t,h) post-processing: v mix, LN(q/k), token shift, RoPE, transpose
// to (B,H,T,N). One wave (32 lanes) per row; lane owns pair (2l, 2l+1).
// ---------------------------------------------------------------------------
__device__ __forceinline__ float2 ln_pair(float a0, float a1, float w0, float w1,
                                          float b0, float b1){
  float m  = waveSum(a0 + a1) * (1.f/64.f);
  float e0 = a0 - m, e1 = a1 - m;
  float var = waveSum(e0*e0 + e1*e1) * (1.f/64.f);
  float rs = rsqrtf(var + 1e-5f);
  return make_float2(e0*rs*w0 + b0, e1*rs*w1 + b1);
}

__global__ void qkv_post(const float* __restrict__ qr, const float* __restrict__ kr,
                         const float* __restrict__ vr, const float* __restrict__ v1,
                         const float* __restrict__ x_q, const float* __restrict__ x_k,
                         const float* __restrict__ x_v, const float* __restrict__ lambp,
                         const float* __restrict__ lnqw, const float* __restrict__ lnqb,
                         const float* __restrict__ lnkw, const float* __restrict__ lnkb,
                         float* __restrict__ qo, float* __restrict__ ko, float* __restrict__ vo)
{
  const int id = blockIdx.x;
  const int h  = id % Hc;
  const int t  = (id / Hc) % Tc;
  const int b  = id / (Hc * Tc);
  const int l  = threadIdx.x;          // pair index 0..31
  const int d0 = 2*l, d1 = 2*l + 1;
  const int tp = (t == 0) ? 0 : (t - 1);

  const size_t base_t = ((size_t)b*Tc + t)*Cc + h*Nc;
  const size_t base_p = ((size_t)b*Tc + tp)*Cc + h*Nc;
  const float lam = lambp[0];

  float q0t = qr[base_t+d0], q1t = qr[base_t+d1];
  float q0p = qr[base_p+d0], q1p = qr[base_p+d1];
  float k0t = kr[base_t+d0], k1t = kr[base_t+d1];
  float k0p = kr[base_p+d0], k1p = kr[base_p+d1];
  float v0t = (1.f-lam)*vr[base_t+d0] + lam*v1[base_t+d0];
  float v1t = (1.f-lam)*vr[base_t+d1] + lam*v1[base_t+d1];
  float v0p = (1.f-lam)*vr[base_p+d0] + lam*v1[base_p+d0];
  float v1p = (1.f-lam)*vr[base_p+d1] + lam*v1[base_p+d1];

  float2 qnt = ln_pair(q0t, q1t, lnqw[d0], lnqw[d1], lnqb[d0], lnqb[d1]);
  float2 qnp = ln_pair(q0p, q1p, lnqw[d0], lnqw[d1], lnqb[d0], lnqb[d1]);
  float2 knt = ln_pair(k0t, k1t, lnkw[d0], lnkw[d1], lnkb[d0], lnkb[d1]);
  float2 knp = ln_pair(k0p, k1p, lnkw[d0], lnkw[d1], lnkb[d0], lnkb[d1]);

  float xq0 = x_q[h*Nc+d0], xq1 = x_q[h*Nc+d1];
  float xk0 = x_k[h*Nc+d0], xk1 = x_k[h*Nc+d1];
  float xv0 = x_v[h*Nc+d0], xv1 = x_v[h*Nc+d1];

  float qs0 = qnt.x + xq0*(qnp.x - qnt.x);
  float qs1 = qnt.y + xq1*(qnp.y - qnt.y);
  float ks0 = knt.x + xk0*(knp.x - knt.x);
  float ks1 = knt.y + xk1*(knp.y - knt.y);
  float vs0 = v0t + xv0*(v0p - v0t);
  float vs1 = v1t + xv1*(v1p - v1t);

  // partial RoPE: pairs 0..15 rotate, pairs 16..31 identity
  float c_ = 1.f, s_ = 0.f;
  if (l < PARTIALc/2) {
    float f  = __expf(-(float)l * (9.2103403720f / 15.f)); // 10000^{-l/15}
    float th = (float)t * f;
    __sincosf(th, &s_, &c_);
  }
  float qo0 = c_*qs0 + s_*qs1;
  float qo1 = c_*qs1 - s_*qs0;
  float ko0 = c_*ks0 + s_*ks1;
  float ko1 = c_*ks1 - s_*ks0;

  const size_t ob = ((size_t)(b*Hc + h)*Tc + t)*Nc;
  qo[ob+d0] = qo0; qo[ob+d1] = qo1;
  ko[ob+d0] = ko0; ko[ob+d1] = ko1;
  vo[ob+d0] = vs0; vo[ob+d1] = vs1;
}

// kw[b,h,t] = sigmoid( dot(x[b,t,:], Wkw[h,:]) )  — one wave per element
__global__ void kw_kernel(const float* __restrict__ x, const float* __restrict__ Wkw,
                          float* __restrict__ kw)
{
  const int id = blockIdx.x;
  const int h  = id % Hc;
  const int t  = (id / Hc) % Tc;
  const int b  = id / (Hc * Tc);
  const int lane = threadIdx.x;
  float s = 0.f;
  const float* xr = x + ((size_t)b*Tc + t)*Cc;
  const float* wr = Wkw + (size_t)h*Cc;
  for (int c = lane; c < Cc; c += 32) s += xr[c] * wr[c];
  s = waveSum(s);
  if (lane == 0) kw[((size_t)b*Hc + h)*Tc + t] = 1.f / (1.f + __expf(-s));
}

// ---------------------------------------------------------------------------
// Sequential chunk scan: one block per (b,h), 256 threads = 8 waves.
// ---------------------------------------------------------------------------
#define SMEM_BYTES (768*64*2*2 + 32*768*4 + 32*4 + 256*4 + 256*4)

__global__ void chunk_kernel(const float* __restrict__ q_r, const float* __restrict__ k_r,
                             const float* __restrict__ v_r, const float* __restrict__ kw,
                             float* __restrict__ d_k, float* __restrict__ d_v,
                             float* __restrict__ d_cnt,
                             const float* __restrict__ lnw, const float* __restrict__ lnb,
                             const float* __restrict__ front_temp,
                             const float* __restrict__ state_temp,
                             float* __restrict__ y)
{
  extern __shared__ char smem_raw[];
  bf16*  s_kstar  = (bf16*)smem_raw;                 // 768 x 64 (unscaled)
  bf16*  s_vstar  = s_kstar + 768*64;                // 768 x 64
  float* s_scores = (float*)(s_vstar + 768*64);      // 32 x 768
  bf16*  s_ck0    = (bf16*)s_scores;                 // 256 x 64 (phase 3, aliases)
  float* s_rsum   = s_scores + 32*768;
  float* s_kw     = s_rsum + 32;
  int*   s_best   = (int*)(s_kw + 256);

  const int bh   = blockIdx.x;
  const int h    = bh % Hc;
  const int tid  = threadIdx.x;
  const int wid  = tid >> 5;
  const int lane = tid & 31;
  const float st    = state_temp[h];
  const float ft    = front_temp[h];
  const float scale = 0.125f;          // 1/sqrt(N=64)

  const size_t qk_base = (size_t)bh * Tc * Nc;
  float* dkp = d_k   + (size_t)bh * SSTATEc * Nc;
  float* dvp = d_v   + (size_t)bh * SSTATEc * Nc;
  float* dcp = d_cnt + (size_t)bh * SSTATEc;
  const float* kwp = kw + (size_t)bh * Tc;

  for (int ci = 0; ci < Tc / CHUNKc; ++ci) {
    const int qb  = ci * CHUNKc;
    const int bbk = (qb >= CHUNKc) ? (qb - CHUNKc) : 0;   // window start
    const int Lw  = qb + CHUNKc - bbk;                    // window length (256 or 512)
    const int L   = SSTATEc + Lw;                         // total keys
    const int nkt = L / 16;

    // ---- phase 1: build k_star / v_star in LDS ----
    for (int row = wid; row < L; row += 8) {
      if (row < SSTATEc) {
        float a0 = dkp[row*64 + lane];
        float a1 = dkp[row*64 + lane + 32];
        float m  = waveSum(a0 + a1) * (1.f/64.f);
        float e0 = a0 - m, e1 = a1 - m;
        float var = waveSum(e0*e0 + e1*e1) * (1.f/64.f);
        float rs = rsqrtf(var + 1e-5f);
        s_kstar[row*64 + lane]      = tobf(e0*rs*lnw[lane]    + lnb[lane]);
        s_kstar[row*64 + lane + 32] = tobf(e1*rs*lnw[lane+32] + lnb[lane+32]);
        float w0 = dvp[row*64 + lane];
        float w1 = dvp[row*64 + lane + 32];
        float nrm = sqrtf(waveSum(w0*w0 + w1*w1));
        float sc  = dcp[row] / fmaxf(nrm, 1e-12f);
        s_vstar[row*64 + lane]      = tobf(w0*sc);
        s_vstar[row*64 + lane + 32] = tobf(w1*sc);
      } else {
        int t = bbk + row - SSTATEc;
        s_kstar[row*64 + lane]      = tobf(k_r[qk_base + (size_t)t*64 + lane]);
        s_kstar[row*64 + lane + 32] = tobf(k_r[qk_base + (size_t)t*64 + lane + 32]);
        s_vstar[row*64 + lane]      = tobf(v_r[qk_base + (size_t)t*64 + lane]);
        s_vstar[row*64 + lane + 32] = tobf(v_r[qk_base + (size_t)t*64 + lane + 32]);
      }
    }
    __syncthreads();

    // ---- phase 2: attention, 8 query groups of 32 rows ----
    for (int qg = 0; qg < 8; ++qg) {
      const int qrow0 = qb + qg*32;

      // hoist q A-fragments: [qsub][kchunk]
      v16bf qa[2][2];
      {
        int mrow = lane & 15;
        int kb0  = (lane < 16) ? 0 : 16;
        for (int qs = 0; qs < 2; ++qs) {
          const float* qrow = q_r + qk_base + (size_t)(qrow0 + qs*16 + mrow)*64;
          for (int kc2 = 0; kc2 < 2; ++kc2)
            for (int j = 0; j < 8; ++j)
              for (int p = 0; p < 2; ++p) {
                int kk = kc2*32 + kb0 + 2*j + p;
                qa[qs][kc2][2*j+p] = tobf(qrow[kk]);
              }
        }
      }
      // scores -> LDS
      for (int tile = wid; tile < 2*nkt; tile += 8) {
        int qs  = tile & 1;
        int kt  = tile >> 1;
        int key = kt*16 + (lane & 15);
        int kbb = (lane < 16) ? 0 : 16;
        v8f acc = {};
        for (int kc2 = 0; kc2 < 2; ++kc2) {
          v16bf bfr;
          for (int j = 0; j < 8; ++j)
            for (int p = 0; p < 2; ++p)
              bfr[2*j+p] = s_kstar[key*64 + kc2*32 + kbb + 2*j + p];
          acc = __builtin_amdgcn_wmma_f32_16x16x32_bf16(false, qa[qs][kc2], false, bfr,
                                                        (short)0, acc, false, false);
        }
        float tmul = ((kt < SSTATEc/16) ? st : ft) * scale;
        int rb = (lane < 16) ? 0 : 8;
        for (int r = 0; r < 8; ++r)
          s_scores[(qs*16 + rb + r)*768 + kt*16 + (lane & 15)] = acc[r] * tmul;
      }
      __syncthreads();

      // softmax (mask applied here), leave probabilities in LDS
      for (int rr = 0; rr < 4; ++rr) {
        int row = wid*4 + rr;
        int qi  = qrow0 + row;
        float m = -3.0e38f;
        for (int j = lane; j < L; j += 32) {
          bool masked = (j >= SSTATEc) && (bbk + j - SSTATEc > qi);
          float s = masked ? NEGc : s_scores[row*768 + j];
          m = fmaxf(m, s);
        }
        m = waveMax(m);
        float sum = 0.f;
        for (int j = lane; j < L; j += 32) {
          bool masked = (j >= SSTATEc) && (bbk + j - SSTATEc > qi);
          float p = masked ? 0.f : __expf(s_scores[row*768 + j] - m);
          s_scores[row*768 + j] = p;
          sum += p;
        }
        sum = waveSum(sum);
        if (lane == 0) s_rsum[row] = sum;
      }
      __syncthreads();

      // out = P @ v_star  (8 output tiles, one per wave)
      {
        int qs = wid & 1;
        int nt = wid >> 1;
        v8f acc = {};
        int mrow = lane & 15;
        int kb0  = (lane < 16) ? 0 : 16;
        int kbb  = (lane < 16) ? 0 : 16;
        for (int kc = 0; kc < L; kc += 32) {
          v16bf pa, bfr;
          for (int j = 0; j < 8; ++j)
            for (int p = 0; p < 2; ++p) {
              int kk = kc + kb0 + 2*j + p;
              pa[2*j+p]  = tobf(s_scores[(qs*16 + mrow)*768 + kk]);
              int kr = kc + kbb + 2*j + p;
              bfr[2*j+p] = s_vstar[kr*64 + nt*16 + (lane & 15)];
            }
          acc = __builtin_amdgcn_wmma_f32_16x16x32_bf16(false, pa, false, bfr,
                                                        (short)0, acc, false, false);
        }
        int rb = (lane < 16) ? 0 : 8;
        for (int r = 0; r < 8; ++r) {
          int rowloc = qs*16 + rb + r;
          float inv = 1.f / s_rsum[rowloc];
          y[qk_base + (size_t)(qrow0 + rowloc)*64 + nt*16 + (lane & 15)] = acc[r] * inv;
        }
      }
      __syncthreads();
    }

    // ---- phase 3: state update ----
    const int cb = bbk;
    for (int row = wid; row < SSTATEc; row += 8) {
      int t = cb + row;
      float a1 = k_r[qk_base + (size_t)t*64 + lane + 32];  // dims <32 zeroed
      float m  = waveSum(a1) * (1.f/64.f);
      float e0 = -m, e1 = a1 - m;
      float var = waveSum(e0*e0 + e1*e1) * (1.f/64.f);
      float rs = rsqrtf(var + 1e-5f);
      s_ck0[row*64 + lane]      = tobf(e0*rs*lnw[lane]    + lnb[lane]);
      s_ck0[row*64 + lane + 32] = tobf(e1*rs*lnw[lane+32] + lnb[lane+32]);
      if (lane == 0) s_kw[row] = kwp[t];
    }
    __syncthreads();

    // sim = ck0 @ d_k_norm^T, running argmax per row
    for (int half = 0; half < 2; ++half) {
      int rt = wid + half*8;
      v16bf ca[2];
      int mrow = lane & 15;
      int kb0  = (lane < 16) ? 0 : 16;
      int kbb  = (lane < 16) ? 0 : 16;
      for (int kc2 = 0; kc2 < 2; ++kc2)
        for (int j = 0; j < 8; ++j)
          for (int p = 0; p < 2; ++p) {
            int kk = kc2*32 + kb0 + 2*j + p;
            ca[kc2][2*j+p] = s_ck0[(rt*16 + mrow)*64 + kk];
          }
      float bv[8]; int bi[8];
      for (int r = 0; r < 8; ++r) { bv[r] = -3.0e38f; bi[r] = 0; }
      for (int ct = 0; ct < SSTATEc/16; ++ct) {
        v8f acc = {};
        int scol = ct*16 + (lane & 15);
        for (int kc2 = 0; kc2 < 2; ++kc2) {
          v16bf bfr;
          for (int j = 0; j < 8; ++j)
            for (int p = 0; p < 2; ++p)
              bfr[2*j+p] = s_kstar[scol*64 + kc2*32 + kbb + 2*j + p];
          acc = __builtin_amdgcn_wmma_f32_16x16x32_bf16(false, ca[kc2], false, bfr,
                                                        (short)0, acc, false, false);
        }
        for (int r = 0; r < 8; ++r) {
          float v = (scol < SINKc) ? NEGc : acc[r];
          if (v > bv[r] || (v == bv[r] && scol < bi[r])) { bv[r] = v; bi[r] = scol; }
        }
      }
      for (int r = 0; r < 8; ++r) {
        float v = bv[r]; int idx = bi[r];
        for (int o = 1; o < 16; o <<= 1) {
          float ov = __shfl_xor(v, o, 32);
          int   oi = __shfl_xor(idx, o, 32);
          if (ov > v || (ov == v && oi < idx)) { v = ov; idx = oi; }
        }
        if ((lane & 15) == 0) {
          int rb = (lane < 16) ? 0 : 8;
          s_best[rt*16 + rb + r] = idx;
        }
      }
    }
    __syncthreads();

    // scatter-add into global state (atomic f32)
    {
      int   c   = tid;
      int   s   = s_best[c];
      float kwv = s_kw[c];
      const float* vrow = v_r + qk_base + (size_t)(cb + c)*64;
      for (int d = 0; d < 64; ++d) {
        atomicAdd(&dkp[s*64 + d], (float)s_ck0[c*64 + d] * kwv);
        atomicAdd(&dvp[s*64 + d], vrow[d] * kwv);
      }
      atomicAdd(&dcp[s], 1.0f);
    }
    __threadfence();
    __syncthreads();
  }
}

// LN over C=1024 of y (transposing (B,H,T,N) -> (B,T,C))
__global__ void yln_kernel(const float* __restrict__ y, const float* __restrict__ w,
                           const float* __restrict__ b, float* __restrict__ out)
{
  const int bt = blockIdx.x;
  const int bq = bt / Tc;
  const int t  = bt % Tc;
  const int tid  = threadIdx.x;
  const int wid  = tid >> 5;
  const int lane = tid & 31;
  __shared__ float red[8];
  __shared__ float s_m, s_v;

  float vals[4];
  float s = 0.f;
  for (int i = 0; i < 4; ++i) {
    int c = tid + i*256;
    int h = c >> 6, d = c & 63;
    vals[i] = y[((size_t)(bq*Hc + h)*Tc + t)*64 + d];
    s += vals[i];
  }
  s = waveSum(s);
  if (lane == 0) red[wid] = s;
  __syncthreads();
  if (tid == 0) { float tot = 0; for (int j = 0; j < 8; ++j) tot += red[j]; s_m = tot * (1.f/1024.f); }
  __syncthreads();
  float m = s_m;
  s = 0.f;
  for (int i = 0; i < 4; ++i) { float e = vals[i] - m; s += e*e; }
  s = waveSum(s);
  __syncthreads();
  if (lane == 0) red[wid] = s;
  __syncthreads();
  if (tid == 0) { float tot = 0; for (int j = 0; j < 8; ++j) tot += red[j]; s_v = tot * (1.f/1024.f); }
  __syncthreads();
  float rs = rsqrtf(s_v + 1e-5f);
  for (int i = 0; i < 4; ++i) {
    int c = tid + i*256;
    out[((size_t)bq*Tc + t)*Cc + c] = (vals[i] - m)*rs*w[c] + b[c];
  }
}

// ---------------------------------------------------------------------------
extern "C" void kernel_launch(void* const* d_in, const int* in_sizes, int n_in,
                              void* d_out, int out_size, void* d_ws, size_t ws_size,
                              hipStream_t stream)
{
  (void)in_sizes; (void)n_in; (void)out_size; (void)ws_size;
  const float* residual  = (const float*)d_in[0];
  const float* x         = (const float*)d_in[1];
  const float* v1        = (const float*)d_in[2];
  const float* Wq        = (const float*)d_in[6];
  const float* Wk        = (const float*)d_in[7];
  const float* Wv        = (const float*)d_in[8];
  const float* Wproj     = (const float*)d_in[9];
  const float* Wkw       = (const float*)d_in[10];
  const float* x_q       = (const float*)d_in[11];
  const float* x_k       = (const float*)d_in[12];
  const float* x_v       = (const float*)d_in[13];
  const float* lambp     = (const float*)d_in[14];
  const float* ln_q_w    = (const float*)d_in[15];
  const float* ln_q_b    = (const float*)d_in[16];
  const float* ln_k_w    = (const float*)d_in[17];
  const float* ln_k_b    = (const float*)d_in[18];
  const float* ln_dk_w   = (const float*)d_in[19];
  const float* ln_dk_b   = (const float*)d_in[20];
  const float* ln_res_w  = (const float*)d_in[21];
  const float* ln_res_b  = (const float*)d_in[22];
  const float* front_temp = (const float*)d_in[23];
  const float* state_temp = (const float*)d_in[24];
  float* out = (float*)d_out;

  float* ws = (float*)d_ws;
  const size_t SZ  = (size_t)Bc*Tc*Cc;                 // 8,388,608
  const size_t BHT = (size_t)Bc*Hc*Tc;                 // 131,072
  const size_t SST = (size_t)Bc*Hc*SSTATEc*Nc;         // 4,194,304
  float* q_raw = ws;
  float* k_raw = ws + SZ;
  float* v_raw = ws + 2*SZ;
  float* q_r   = ws + 3*SZ;
  float* k_r   = ws + 4*SZ;
  float* v_r   = ws + 5*SZ;
  float* kw    = ws + 6*SZ;
  float* d_k   = kw + BHT;
  float* d_v   = d_k + SST;
  float* d_cnt = d_v + SST;
  float* y     = q_raw;   // raw buffers are dead after qkv_post
  float* yln   = k_raw;

  dim3 gridG(Mrows/64, Cc/64);
  gemm_wmma<<<gridG, 128, 0, stream>>>(x, Wq, nullptr, q_raw, Mrows, Cc, Cc);
  gemm_wmma<<<gridG, 128, 0, stream>>>(x, Wk, nullptr, k_raw, Mrows, Cc, Cc);
  gemm_wmma<<<gridG, 128, 0, stream>>>(x, Wv, nullptr, v_raw, Mrows, Cc, Cc);

  qkv_post<<<Bc*Tc*Hc, 32, 0, stream>>>(q_raw, k_raw, v_raw, v1,
                                        x_q, x_k, x_v, lambp,
                                        ln_q_w, ln_q_b, ln_k_w, ln_k_b,
                                        q_r, k_r, v_r);
  kw_kernel<<<Bc*Tc*Hc, 32, 0, stream>>>(x, Wkw, kw);

  hipMemsetAsync((void*)d_k, 0, (2*SST + (size_t)Bc*Hc*SSTATEc)*sizeof(float), stream);

  hipFuncSetAttribute((const void*)chunk_kernel,
                      hipFuncAttributeMaxDynamicSharedMemorySize, SMEM_BYTES);
  chunk_kernel<<<Bc*Hc, 256, SMEM_BYTES, stream>>>(q_r, k_r, v_r, kw,
                                                   d_k, d_v, d_cnt,
                                                   ln_dk_w, ln_dk_b,
                                                   front_temp, state_temp, y);

  yln_kernel<<<Bc*Tc, 256, 0, stream>>>(y, ln_res_w, ln_res_b, yln);

  gemm_wmma<<<gridG, 128, 0, stream>>>(yln, Wproj, residual, out, Mrows, Cc, Cc);
}
